// MultiHeadAttentionOne_20074677142286
// MI455X (gfx1250) — matrix operand order repaired and checked by reading, not compiled, not run on hardware
//
#include <hip/hip_runtime.h>

#define SEQ_L 4096
#define HD 64
#define DMODEL 256
#define NB_PAD 80   // padded row stride (halfwords) for 16B-aligned, bank-spread LDS tiles

typedef __attribute__((ext_vector_type(16))) __bf16 v16bf;
typedef __attribute__((ext_vector_type(8)))  float v8f;
typedef __attribute__((ext_vector_type(4)))  unsigned int v4u;

union BFrag {
    v16bf v;
    unsigned short s[16];
    v4u q[2];
};

union U4S {
    v4u u;
    unsigned short s[8];
};

__device__ __forceinline__ unsigned short f2bf(float f) {
    unsigned int u = __float_as_uint(f);
    unsigned int r = u + 0x7FFFu + ((u >> 16) & 1u);  // round-to-nearest-even
    return (unsigned short)(r >> 16);
}

// ---------------------------------------------------------------------------
// Phase 1: qkv[b,l,j] = sum_c q[b,c,l] * w_qkv[j,c]; store bf16 qh[h][b][l][d]
// Block: 256 threads, handles 64 l-values for one b. LDS: x tile + w chunk.
// ---------------------------------------------------------------------------
__global__ __launch_bounds__(256) void qkv_proj(const float* __restrict__ q,
                                                const float* __restrict__ w_qkv,
                                                unsigned short* __restrict__ qh) {
    __shared__ float xs[64 * 257];   // xs[l][c]
    __shared__ float ws[64 * 257];   // ws[jloc][c]
    const int blk = blockIdx.x;
    const int b = blk >> 6;
    const int l0 = (blk & 63) * 64;
    const int tid = threadIdx.x;

    // load x tile: coalesced along l (64 consecutive threads share c)
    {
        const int lx = tid & 63, cb = tid >> 6;
        for (int i = 0; i < 64; ++i) {
            const int c = cb * 64 + i;
            xs[lx * 257 + c] = q[(size_t)(b * DMODEL + c) * SEQ_L + l0 + lx];
        }
    }
    const int jloc = tid & 63;
    const int lgrp = tid >> 6;

    for (int jc = 0; jc < 4; ++jc) {   // 4 chunks of 64 output channels == head jc
        // load w chunk (rows jc*64 .. +63), coalesced
        for (int i = 0; i < 64; ++i) {
            const int idx = tid + 256 * i;
            const int r = idx >> 8, cc = idx & 255;
            ws[r * 257 + cc] = w_qkv[(size_t)(jc * 64 + r) * DMODEL + cc];
        }
        __syncthreads();

        float acc[16];
#pragma unroll
        for (int i = 0; i < 16; ++i) acc[i] = 0.f;
        for (int c = 0; c < DMODEL; ++c) {
            const float wv = ws[jloc * 257 + c];
#pragma unroll
            for (int i = 0; i < 16; ++i)
                acc[i] += xs[(lgrp * 16 + i) * 257 + c] * wv;
        }
        // store bf16: qh[((h*2+b)*L + l)*64 + d], h=jc, d=jloc
#pragma unroll
        for (int i = 0; i < 16; ++i) {
            const int l = l0 + lgrp * 16 + i;
            qh[((size_t)(jc * 2 + b) * SEQ_L + l) * HD + jloc] = f2bf(acc[i]);
        }
        __syncthreads();
    }
}

// ---------------------------------------------------------------------------
// Phase 2: flash attention, bf16 WMMA, fp32 accumulate.
// Grid: (64 query tiles, 8 head-batch). 128 threads = 4 waves; each wave owns
// a 16-row query tile and 16x64 fp32 O accumulator.
// ---------------------------------------------------------------------------
__global__ __launch_bounds__(128) void flash_attn(const unsigned short* __restrict__ qh,
                                                  float* __restrict__ o_ws) {
    __shared__ alignas(16) unsigned short kt[64 * NB_PAD];       // K rows  [key][d]
    __shared__ alignas(16) unsigned short vt[64 * NB_PAD];       // V^T     [d][key]
    __shared__ alignas(16) unsigned short pls[4 * 16 * NB_PAD];  // per-wave P [m][key]

    const int hb = blockIdx.y;            // h*2 + b
    const int qt = blockIdx.x;            // query tile of 64
    const int tid = threadIdx.x;
    const int wave = tid >> 5, lane = tid & 31;
    const int m16 = lane & 15, hl = lane >> 4;
    const unsigned short* base = qh + (size_t)hb * SEQ_L * HD;

    // Q as two A-fragments (d chunks of 32). A layout: row m = lane&15;
    // lane holds K-groups [32c+8h .. +7] and [32c+16+8h .. +7].
    BFrag aq[2];
    const int qrow = qt * 64 + wave * 16 + m16;
#pragma unroll
    for (int c = 0; c < 2; ++c) {
        const unsigned short* p = base + (size_t)qrow * HD + c * 32 + hl * 8;
        aq[c].q[0] = *(const v4u*)(p);
        aq[c].q[1] = *(const v4u*)(p + 16);
    }

    v8f of[4];
    float rmax[8], rsum[8];
#pragma unroll
    for (int nt = 0; nt < 4; ++nt)
#pragma unroll
        for (int e = 0; e < 8; ++e) of[nt][e] = 0.f;
#pragma unroll
    for (int v = 0; v < 8; ++v) { rmax[v] = -1.0e30f; rsum[v] = 0.f; }

    unsigned short* pw = &pls[wave * 16 * NB_PAD];

    for (int kb = 0; kb < SEQ_L / 64; ++kb) {
        // stage K tile (row-major) and V^T tile (same data; q==k==v)
#pragma unroll
        for (int ii = 0; ii < 4; ++ii) {
            const int idx = tid + 128 * ii;              // 0..511
            const int row = idx >> 3, col8 = idx & 7;
            U4S d;
            d.u = *(const v4u*)(base + (size_t)(kb * 64 + row) * HD + col8 * 8);
            *(v4u*)&kt[row * NB_PAD + col8 * 8] = d.u;
#pragma unroll
            for (int e = 0; e < 8; ++e)
                vt[(col8 * 8 + e) * NB_PAD + row] = d.s[e];
        }
        __syncthreads();

        // S = Q K^T  (4 n-tiles x 2 k-chunks = 8 WMMAs)
        v8f sc[4];
#pragma unroll
        for (int nt = 0; nt < 4; ++nt) {
            v8f c;
#pragma unroll
            for (int e = 0; e < 8; ++e) c[e] = 0.f;
#pragma unroll
            for (int kc = 0; kc < 2; ++kc) {
                BFrag bk;  // B[k][n] = Key[n][k]; lane n holds d groups [8h..+7] and [16+8h..+7]
                const unsigned short* p = &kt[(nt * 16 + m16) * NB_PAD + kc * 32 + hl * 8];
                bk.q[0] = *(const v4u*)(p);
                bk.q[1] = *(const v4u*)(p + 16);
                c = __builtin_amdgcn_wmma_f32_16x16x32_bf16(
                        false, aq[kc].v, false, bk.v, (short)0, c, false, false);
            }
            sc[nt] = c;
        }

        // online softmax per row (row v+8*hl lives in c[v] across 16 lanes)
#pragma unroll
        for (int v = 0; v < 8; ++v) {
            const float s0 = sc[0][v] * 0.125f;
            const float s1 = sc[1][v] * 0.125f;
            const float s2 = sc[2][v] * 0.125f;
            const float s3 = sc[3][v] * 0.125f;
            float t = fmaxf(fmaxf(s0, s1), fmaxf(s2, s3));
#pragma unroll
            for (int m = 1; m <= 8; m <<= 1) t = fmaxf(t, __shfl_xor(t, m));
            const float mnew = fmaxf(rmax[v], t);
            const float corr = __expf(rmax[v] - mnew);
            const float p0 = __expf(s0 - mnew), p1 = __expf(s1 - mnew);
            const float p2 = __expf(s2 - mnew), p3 = __expf(s3 - mnew);
            float rs = p0 + p1 + p2 + p3;
#pragma unroll
            for (int m = 1; m <= 8; m <<= 1) rs += __shfl_xor(rs, m);
            rsum[v] = rsum[v] * corr + rs;
            rmax[v] = mnew;
#pragma unroll
            for (int nt = 0; nt < 4; ++nt) of[nt][v] = of[nt][v] * corr;
            const int r = v + 8 * hl;
            pw[r * NB_PAD + 0 * 16 + m16] = f2bf(p0);
            pw[r * NB_PAD + 1 * 16 + m16] = f2bf(p1);
            pw[r * NB_PAD + 2 * 16 + m16] = f2bf(p2);
            pw[r * NB_PAD + 3 * 16 + m16] = f2bf(p3);
        }
        // wave-private P staging: only need own DS ops complete
        asm volatile("s_wait_dscnt 0" ::: "memory");

        // O += P V  (8 WMMAs)
#pragma unroll
        for (int kc = 0; kc < 2; ++kc) {
            BFrag ap;
            const unsigned short* p = &pw[m16 * NB_PAD + kc * 32 + hl * 8];
            ap.q[0] = *(const v4u*)(p);
            ap.q[1] = *(const v4u*)(p + 16);
#pragma unroll
            for (int nt = 0; nt < 4; ++nt) {
                BFrag bv;  // B[k][n] = V[key k][d n] = vt[n][k]
                const unsigned short* pv = &vt[(nt * 16 + m16) * NB_PAD + kc * 32 + hl * 8];
                bv.q[0] = *(const v4u*)(pv);
                bv.q[1] = *(const v4u*)(pv + 16);
                of[nt] = __builtin_amdgcn_wmma_f32_16x16x32_bf16(
                             false, ap.v, false, bv.v, (short)0, of[nt], false, false);
            }
        }
        __syncthreads();
    }

    // epilogue: O / rowsum  ->  o_ws[b][l][h*64 + d]  (fp32)
    const int h = hb >> 1, b = hb & 1;
#pragma unroll
    for (int v = 0; v < 8; ++v) {
        const float inv = 1.0f / rsum[v];
        const int l = qt * 64 + wave * 16 + v + 8 * hl;
        float* orow = o_ws + ((size_t)(b * SEQ_L + l)) * DMODEL + h * HD;
#pragma unroll
        for (int nt = 0; nt < 4; ++nt)
            orow[nt * 16 + m16] = of[nt][v] * inv;
    }
}

// ---------------------------------------------------------------------------
// Phase 3: y = o @ fc_w^T + fc_b + residual; LayerNorm; write [b][l][c] flat.
// Block: 256 threads, 32 l-values per block.
// ---------------------------------------------------------------------------
__global__ __launch_bounds__(256) void fc_ln(const float* __restrict__ o_ws,
                                             const float* __restrict__ fc_w,
                                             const float* __restrict__ fc_b,
                                             const float* __restrict__ ln_g,
                                             const float* __restrict__ ln_b,
                                             const float* __restrict__ q,
                                             float* __restrict__ out) {
    __shared__ float os[32 * 257];   // attention output tile [l][k]
    __shared__ float ws[64 * 257];   // fc_w chunk [jloc][k]
    __shared__ float ys[32 * 257];   // accumulator [l][j]
    const int blk = blockIdx.x;
    const int b = blk >> 7;
    const int l0 = (blk & 127) * 32;
    const int tid = threadIdx.x;

    // load o tile (contiguous -> coalesced)
    {
        const size_t gbase = ((size_t)(b * SEQ_L + l0)) * DMODEL;
        for (int i = 0; i < 32; ++i) {
            const int idx = tid + 256 * i;
            os[(idx >> 8) * 257 + (idx & 255)] = o_ws[gbase + idx];
        }
    }
    // init ys with residual + bias: ys[l][j] = q[b][j][l0+l] + fc_b[j]
    {
        const int lx = tid & 31, cb = tid >> 5;   // cb in 0..7
        for (int i = 0; i < 32; ++i) {
            const int j = cb * 32 + i;
            ys[lx * 257 + j] = q[(size_t)(b * DMODEL + j) * SEQ_L + l0 + lx] + fc_b[j];
        }
    }

    const int jloc = tid & 63;
    const int lgrp = tid >> 6;   // 0..3, 8 rows each
    for (int jc = 0; jc < 4; ++jc) {
        for (int i = 0; i < 64; ++i) {
            const int idx = tid + 256 * i;
            const int r = idx >> 8, cc = idx & 255;
            ws[r * 257 + cc] = fc_w[(size_t)(jc * 64 + r) * DMODEL + cc];
        }
        __syncthreads();

        float acc[8];
#pragma unroll
        for (int i = 0; i < 8; ++i) acc[i] = 0.f;
        for (int c = 0; c < DMODEL; ++c) {
            const float wv = ws[jloc * 257 + c];
#pragma unroll
            for (int i = 0; i < 8; ++i)
                acc[i] += os[(lgrp * 8 + i) * 257 + c] * wv;
        }
        const int j = jc * 64 + jloc;
#pragma unroll
        for (int i = 0; i < 8; ++i)
            ys[(lgrp * 8 + i) * 257 + j] += acc[i];
        __syncthreads();
    }

    // LayerNorm across j (256) per row; 8 waves x 4 rows
    const int wave = tid >> 5, lane = tid & 31;
    for (int k = 0; k < 4; ++k) {
        const int r = wave * 4 + k;
        float s = 0.f, ss = 0.f;
#pragma unroll
        for (int g = 0; g < 8; ++g) {
            const float v = ys[r * 257 + lane + 32 * g];
            s += v; ss += v * v;
        }
#pragma unroll
        for (int m = 1; m <= 16; m <<= 1) {
            s += __shfl_xor(s, m);
            ss += __shfl_xor(ss, m);
        }
        const float mu = s * (1.0f / 256.0f);
        const float var = ss * (1.0f / 256.0f) - mu * mu;
        const float rstd = rsqrtf(var + 1e-5f);
        float* orow = out + ((size_t)(b * SEQ_L + l0 + r)) * DMODEL;
#pragma unroll
        for (int g = 0; g < 8; ++g) {
            const int j = lane + 32 * g;
            const float v = ys[r * 257 + j];
            orow[j] = (v - mu) * rstd * ln_g[j] + ln_b[j];
        }
    }
}

extern "C" void kernel_launch(void* const* d_in, const int* in_sizes, int n_in,
                              void* d_out, int out_size, void* d_ws, size_t ws_size,
                              hipStream_t stream) {
    const float* q     = (const float*)d_in[0];
    const float* w_qkv = (const float*)d_in[1];
    const float* fc_w  = (const float*)d_in[2];
    const float* fc_b  = (const float*)d_in[3];
    const float* ln_g  = (const float*)d_in[4];
    const float* ln_b  = (const float*)d_in[5];
    float* out = (float*)d_out;

    // workspace: qh bf16 [4][2][4096][64] = 4 MB, then o fp32 [2][4096][256] = 8 MB
    unsigned short* qh = (unsigned short*)d_ws;
    float* o_ws = (float*)((char*)d_ws + (size_t)4 * 1024 * 1024);

    qkv_proj<<<128, 256, 0, stream>>>(q, w_qkv, qh);
    flash_attn<<<dim3(64, 8), 128, 0, stream>>>(qh, o_ws);
    fc_ln<<<256, 256, 0, stream>>>(o_ws, fc_w, fc_b, ln_g, ln_b, q, out);
}
